// Frame_39118562132116
// MI455X (gfx1250) — matrix operand order repaired and checked by reading, not compiled, not run on hardware
//
#include <hip/hip_runtime.h>
#include <hip/hip_bf16.h>

#define S_ 8
#define N_ 1024
#define K_ 8
#define D_ 3
#define L_ 64
#define H_ 256
#define SN (S_*N_)                 /* 8192  */
#define SNKD (S_*N_*K_*D_)         /* 196608 */
#define PTS_PER_SAMPLE (N_*K_*D_)  /* 24576 */
#define MTOT (SN + 2*SNKD)         /* 401408 -> 12544 tiles of 32 */
#define MT 32                      /* points per workgroup (two 16-row WMMA tiles) */
#define K1PAD 96
#define EPSF 1e-7f

typedef __attribute__((ext_vector_type(16))) _Float16 v16h;
typedef __attribute__((ext_vector_type(8)))  _Float16 v8h;
typedef __attribute__((ext_vector_type(8)))  float    v8f;

static __device__ __forceinline__ v16h cat8(v8h lo, v8h hi) {
  v16h r;
#pragma unroll
  for (int i = 0; i < 8; ++i) { r[i] = lo[i]; r[i + 8] = hi[i]; }
  return r;
}

// ---------------------------------------------------------------------------
// Kernel 0: convert fp32 weights -> f16, transposed (row = output column n,
// 16B-vectorizable along k). W1 zero-padded 67 -> 96 along k.
// ---------------------------------------------------------------------------
__global__ void wconv_kernel(const float* __restrict__ W1,
                             const float* __restrict__ W2,
                             const float* __restrict__ W3,
                             _Float16* __restrict__ W1hT,
                             _Float16* __restrict__ W2hT,
                             _Float16* __restrict__ W3h) {
  int i = blockIdx.x * blockDim.x + threadIdx.x;
  if (i < H_ * K1PAD) {
    int n = i / K1PAD, k = i - n * K1PAD;
    float v = (k < D_ + L_) ? W1[k * H_ + n] : 0.0f;
    W1hT[i] = (_Float16)v;
  } else if (i < H_ * K1PAD + H_ * H_) {
    int j = i - H_ * K1PAD;
    int n = j / H_, k = j - n * H_;
    W2hT[j] = (_Float16)W2[k * H_ + n];
  } else if (i < H_ * K1PAD + H_ * H_ + H_) {
    int k = i - (H_ * K1PAD + H_ * H_);
    W3h[k] = (_Float16)W3[k];
  }
}

// ---------------------------------------------------------------------------
// Kernel 1: fused 3-layer MLP over one 32-point tile per workgroup.
// 128 threads = 4 wave32; each wave owns 4 of the 16 N-tiles and both 16-row
// M-tiles, so every B (weight) load feeds two v_wmma. Layouts per ISA 7.12.2.
// ---------------------------------------------------------------------------
__global__ __launch_bounds__(128)
void sdf_mlp_kernel(const float* __restrict__ x, const float* __restrict__ slat,
                    const float* __restrict__ rf,
                    const _Float16* __restrict__ W1hT,
                    const _Float16* __restrict__ W2hT,
                    const _Float16* __restrict__ W3h,
                    const float* __restrict__ b1, const float* __restrict__ b2,
                    const float* __restrict__ b3,
                    float* __restrict__ fvals) {
  __shared__ _Float16 feat[MT][K1PAD] __attribute__((aligned(32)));
  __shared__ _Float16 h1[MT][H_]      __attribute__((aligned(32)));
  __shared__ _Float16 h2[MT][H_]      __attribute__((aligned(32)));
  __shared__ float    part[MT][4];

  const int tid  = threadIdx.x;
  const int lane = tid & 31;
  const int wave = tid >> 5;
  const int q0   = blockIdx.x * MT;

  __builtin_prefetch(W2hT, 0, 1);  // global_prefetch_b8: pull weights toward L2

  // ---- stage 32 feature rows [pt(3) | latent(64) | pad..96] as f16 in LDS
  for (int j = tid; j < MT * K1PAD; j += 128) {
    int row = j / K1PAD;
    int col = j - row * K1PAD;
    float v = 0.0f;
    if (col < D_ + L_) {
      int q = q0 + row;
      int sample, n, k = 0, d = 0, sgn = 0;
      if (q < SN) {
        sample = q >> 10; n = q & (N_ - 1);
      } else {
        int r = q - SN;
        sgn = 1;
        if (r >= SNKD) { r -= SNKD; sgn = -1; }
        sample = r / PTS_PER_SAMPLE;
        int w = r - sample * PTS_PER_SAMPLE;
        n = w / (K_ * D_);
        int rem = w - n * (K_ * D_);
        k = rem / D_; d = rem - k * D_;
      }
      if (col < D_) {
        v = x[(sample * N_ + n) * D_ + col];
        if (sgn)
          v += (float)sgn * rf[((((sample * N_ + n) * K_) + k) * D_ + d) * D_ + col];
      } else {
        v = slat[sample * L_ + (col - D_)];
      }
    }
    feat[row][col] = (_Float16)v;
  }
  __syncthreads();

  const int mrow = lane & 15;
  const int kbA  = (lane < 16) ? 0 : 8;   // A-matrix per-lane K base (ISA 7.12.2)
  const int kbB  = (lane < 16) ? 0 : 16;  // unused; B loads use kbA as well
  const int ncl  = lane & 15;

  // ---- layer 1: [32 x 96] x [96 x 256], 3 WMMA k-steps, 2 M-tiles per B load
#pragma unroll
  for (int t = 0; t < 4; ++t) {
    const int nt = wave * 4 + t;
    const int ncol = nt * 16 + ncl;
    v8f c0, c1;
    const float bv = b1[ncol];
#pragma unroll
    for (int i = 0; i < 8; ++i) { c0[i] = bv; c1[i] = bv; }
#pragma unroll
    for (int ks = 0; ks < 3; ++ks) {
      const int k0 = ks * 32;
      v16h a0 = cat8(*(const v8h*)&feat[mrow][k0 + kbA],
                     *(const v8h*)&feat[mrow][k0 + 16 + kbA]);
      v16h a1 = cat8(*(const v8h*)&feat[mrow + 16][k0 + kbA],
                     *(const v8h*)&feat[mrow + 16][k0 + 16 + kbA]);
      const _Float16* wp = &W1hT[ncol * K1PAD + k0 + kbA];
      v16h bm = cat8(*(const v8h*)wp, *(const v8h*)(wp + 16));
      c0 = __builtin_amdgcn_wmma_f32_16x16x32_f16(false, a0, false, bm,
                                                  (short)0, c0, false, false);
      c1 = __builtin_amdgcn_wmma_f32_16x16x32_f16(false, a1, false, bm,
                                                  (short)0, c1, false, false);
    }
#pragma unroll
    for (int v = 0; v < 8; ++v) {
      const int row = (lane < 16) ? v : v + 8;
      const float r0 = c0[v] > 0.0f ? c0[v] : 0.0f;  // ReLU
      const float r1 = c1[v] > 0.0f ? c1[v] : 0.0f;
      h1[row][ncol]      = (_Float16)r0;
      h1[row + 16][ncol] = (_Float16)r1;
    }
  }
  __syncthreads();

  // ---- layer 2: [32 x 256] x [256 x 256], 8 WMMA k-steps, 2 M-tiles per B load
#pragma unroll
  for (int t = 0; t < 4; ++t) {
    const int nt = wave * 4 + t;
    const int ncol = nt * 16 + ncl;
    v8f c0, c1;
    const float bv = b2[ncol];
#pragma unroll
    for (int i = 0; i < 8; ++i) { c0[i] = bv; c1[i] = bv; }
#pragma unroll
    for (int ks = 0; ks < 8; ++ks) {
      const int k0 = ks * 32;
      v16h a0 = cat8(*(const v8h*)&h1[mrow][k0 + kbA],
                     *(const v8h*)&h1[mrow][k0 + 16 + kbA]);
      v16h a1 = cat8(*(const v8h*)&h1[mrow + 16][k0 + kbA],
                     *(const v8h*)&h1[mrow + 16][k0 + 16 + kbA]);
      const _Float16* wp = &W2hT[ncol * H_ + k0 + kbA];
      v16h bm = cat8(*(const v8h*)wp, *(const v8h*)(wp + 16));
      c0 = __builtin_amdgcn_wmma_f32_16x16x32_f16(false, a0, false, bm,
                                                  (short)0, c0, false, false);
      c1 = __builtin_amdgcn_wmma_f32_16x16x32_f16(false, a1, false, bm,
                                                  (short)0, c1, false, false);
    }
#pragma unroll
    for (int v = 0; v < 8; ++v) {
      const int row = (lane < 16) ? v : v + 8;
      const float r0 = c0[v] > 0.0f ? c0[v] : 0.0f;  // ReLU
      const float r1 = c1[v] > 0.0f ? c1[v] : 0.0f;
      h2[row][ncol]      = (_Float16)r0;
      h2[row + 16][ncol] = (_Float16)r1;
    }
  }
  __syncthreads();

  // ---- layer 3: [32 x 256] x [256 x 1] — all 128 threads: each takes a
  // 64-element quarter of one point, then a tiny LDS reduction.
  {
    const int pt = tid >> 2;          // point 0..31
    const int qt = tid & 3;           // quarter 0..3
    float acc = 0.0f;
    const int kb = qt * 64;
#pragma unroll 8
    for (int k = 0; k < 64; ++k)
      acc += (float)h2[pt][kb + k] * (float)W3h[kb + k];
    part[pt][qt] = acc;
  }
  __syncthreads();
  if (tid < MT) {
    fvals[q0 + tid] = part[tid][0] + part[tid][1] + part[tid][2] + part[tid][3]
                    + b3[0];
  }
}

// ---------------------------------------------------------------------------
// Kernel 2: per-(s,n) epilogue — gradient, tangent frame, FD Hessian,
// projected 2x2 symmetric eig, |lambda|-sort + sign fix, frame rotation.
// ---------------------------------------------------------------------------
__global__ __launch_bounds__(128)
void finish_kernel(const float* __restrict__ rfg, const float* __restrict__ bivec,
                   const float* __restrict__ fvals, float* __restrict__ out) {
  int p = blockIdx.x * blockDim.x + threadIdx.x;
  if (p >= SN) return;
  const float* rf = &rfg[p * (K_ * D_ * D_)];
  const float* fp = &fvals[SN + p * (K_ * D_)];
  const float* fm = &fvals[SN + SNKD + p * (K_ * D_)];
  const float fc = fvals[p];

  float grad[3] = {0.f, 0.f, 0.f};
  float Hm[3][3] = {{0.f,0.f,0.f},{0.f,0.f,0.f},{0.f,0.f,0.f}};
#pragma unroll
  for (int k = 0; k < K_; ++k) {
#pragma unroll
    for (int d = 0; d < D_; ++d) {
      const float r0 = rf[(k * D_ + d) * D_ + 0];
      const float r1 = rf[(k * D_ + d) * D_ + 1];
      const float r2 = rf[(k * D_ + d) * D_ + 2];
      const float n2 = r0 * r0 + r1 * r1 + r2 * r2 + EPSF;
      const float ip = fp[k * D_ + d], im = fm[k * D_ + d];
      const float df = (ip - im) / n2;
      grad[0] += df * r0; grad[1] += df * r1; grad[2] += df * r2;
      const float d2 = (ip - 2.0f * fc + im) / (n2 * n2);
      Hm[0][0] += d2 * r0 * r0; Hm[0][1] += d2 * r0 * r1; Hm[0][2] += d2 * r0 * r2;
      Hm[1][1] += d2 * r1 * r1; Hm[1][2] += d2 * r1 * r2; Hm[2][2] += d2 * r2 * r2;
    }
  }
  Hm[1][0] = Hm[0][1]; Hm[2][0] = Hm[0][2]; Hm[2][1] = Hm[1][2];

  float gn = sqrtf(grad[0]*grad[0] + grad[1]*grad[1] + grad[2]*grad[2]) + EPSF;
  grad[0] /= gn; grad[1] /= gn; grad[2] /= gn;

  const float bx = bivec[p*3+0], by = bivec[p*3+1], bz = bivec[p*3+2];
  float crx = grad[1]*bz - grad[2]*by;
  float cry = grad[2]*bx - grad[0]*bz;
  float crz = grad[0]*by - grad[1]*bx;
  float cn = sqrtf(crx*crx + cry*cry + crz*crz) + EPSF;
  crx /= cn; cry /= cn; crz /= cn;
  const float c2x = grad[1]*crz - grad[2]*cry;
  const float c2y = grad[2]*crx - grad[0]*crz;
  const float c2z = grad[0]*cry - grad[1]*crx;

  const float ft[2][3] = {{crx, cry, crz}, {c2x, c2y, c2z}};
  float hp[2][2];
#pragma unroll
  for (int a = 0; a < 2; ++a)
#pragma unroll
    for (int b = 0; b < 2; ++b) {
      float s = 0.f;
#pragma unroll
      for (int i = 0; i < 3; ++i)
#pragma unroll
        for (int j = 0; j < 3; ++j)
          s += ft[a][i] * Hm[i][j] * ft[b][j];
      hp[a][b] = s;
    }

  const float Aq = hp[0][0], Bq = 0.5f * (hp[0][1] + hp[1][0]), Cq = hp[1][1];
  const float mid  = 0.5f * (Aq + Cq);
  const float disc = sqrtf(0.25f * (Aq - Cq) * (Aq - Cq) + Bq * Bq);
  const float l0 = mid - disc, l1 = mid + disc;  // ascending (eigh order)

  auto evec = [](float aa, float bb, float cc, float lam, float& vx, float& vy) {
    vx = bb; vy = lam - aa;
    float n2 = vx * vx + vy * vy;
    if (n2 < 1e-30f) { vx = lam - cc; vy = bb; n2 = vx * vx + vy * vy; }
    if (n2 < 1e-30f) { vx = 1.0f; vy = 0.0f; n2 = 1.0f; }
    const float inv = rsqrtf(n2); vx *= inv; vy *= inv;
  };
  float v0x, v0y, v1x, v1y;
  evec(Aq, Bq, Cq, l0, v0x, v0y);
  evec(Aq, Bq, Cq, l1, v1x, v1y);

  // argsort(-|lambda|) (stable), then sign = 1 - 2*idx[0] applied to column 0
  float e0x, e0y, e1x, e1y, sign;
  if (fabsf(l0) >= fabsf(l1)) { e0x = v0x; e0y = v0y; e1x = v1x; e1y = v1y; sign =  1.0f; }
  else                        { e0x = v1x; e0y = v1y; e1x = v0x; e1y = v0y; sign = -1.0f; }
  e0x *= sign; e0y *= sign;

  // new_tangent[a][d] = sum_b E[b][a] * ft[b][d]
  float* o = &out[p * 9];
  o[0] = grad[0]; o[1] = grad[1]; o[2] = grad[2];
#pragma unroll
  for (int d = 0; d < 3; ++d) {
    o[3 + d] = e0x * ft[0][d] + e0y * ft[1][d];
    o[6 + d] = e1x * ft[0][d] + e1y * ft[1][d];
  }
}

// ---------------------------------------------------------------------------
extern "C" void kernel_launch(void* const* d_in, const int* in_sizes, int n_in,
                              void* d_out, int out_size, void* d_ws, size_t ws_size,
                              hipStream_t stream) {
  const float* x    = (const float*)d_in[0];
  const float* slat = (const float*)d_in[1];
  const float* rf   = (const float*)d_in[2];
  const float* bv   = (const float*)d_in[3];
  const float* W1   = (const float*)d_in[4];
  const float* b1   = (const float*)d_in[5];
  const float* W2   = (const float*)d_in[6];
  const float* b2   = (const float*)d_in[7];
  const float* W3   = (const float*)d_in[8];
  const float* b3   = (const float*)d_in[9];
  (void)in_sizes; (void)n_in; (void)out_size; (void)ws_size;

  char* ws = (char*)d_ws;
  _Float16* W1hT = (_Float16*)ws;                           // 256*96*2  = 49152 B
  _Float16* W2hT = (_Float16*)(ws + 49152);                 // 256*256*2 = 131072 B
  _Float16* W3h  = (_Float16*)(ws + 49152 + 131072);        // 512 B (pad to 1024)
  float*    fvals = (float*)(ws + 49152 + 131072 + 1024);   // MTOT*4 = 1605632 B

  const int convN = H_ * K1PAD + H_ * H_ + H_;              // 90368
  wconv_kernel<<<(convN + 255) / 256, 256, 0, stream>>>(W1, W2, W3, W1hT, W2hT, W3h);
  sdf_mlp_kernel<<<MTOT / MT, 128, 0, stream>>>(x, slat, rf, W1hT, W2hT, W3h,
                                                b1, b2, b3, fvals);
  finish_kernel<<<(SN + 127) / 128, 128, 0, stream>>>(rf, bv, fvals, (float*)d_out);
}
